// SparseNeuralConv_13460427505900
// MI455X (gfx1250) — matrix operand order repaired and checked by reading, not compiled, not run on hardware
//
#include <hip/hip_runtime.h>
#include <hip/hip_bf16.h>
#include <math.h>

// ---------------------------------------------------------------------------
// CDNA5 (gfx1250) wave32 WMMA types
// ---------------------------------------------------------------------------
typedef __attribute__((ext_vector_type(16))) _Float16 v16h;
typedef __attribute__((ext_vector_type(8)))  _Float16 v8h;
typedef __attribute__((ext_vector_type(8)))  float    v8f;

// Problem constants (fixed by reference):
//   n=16, ch=96, res=96, k=3 -> L=1024/img, NL=16384 patches
//   vc=512 codebook rows, MLP dims 105->288->96->144->288, conv 32->96
#define CH     96
#define VC     512
#define NL_TOT 16384
#define PB     16          // patches per workgroup
#define NCOLS  (PB * 9)    // 144 (patch,pos) columns per workgroup

// ---------------------------------------------------------------------------
// Prep: 1/||w_mask_row||
// ---------------------------------------------------------------------------
__global__ void k_wnorm(const float* __restrict__ w, float* __restrict__ rw) {
  int v = blockIdx.x * blockDim.x + threadIdx.x;
  if (v < VC) {
    float s = 0.f;
    for (int c = 0; c < CH; ++c) { float t = w[v * CH + c]; s += t * t; }
    rw[v] = rsqrtf(s);
  }
}

// ---------------------------------------------------------------------------
// Prep: pack row-normalized w_mask into A-fragment layout (f16).
// A tile = 16(M=vc rows) x 32(K=ch). 16-bit A k order used here (contiguous, same as B in k_mask):
//   lanes 0-15 : halves e -> K=e
//   lanes 16-31: halves e -> K=16+e
// Fragment (t, kc) stored as 32 lanes x 16 halves contiguous.
// ---------------------------------------------------------------------------
__global__ void k_pack_wmaskA(const float* __restrict__ w,
                              const float* __restrict__ rw,
                              _Float16* __restrict__ dst) {
  int gid = blockIdx.x * blockDim.x + threadIdx.x;   // total 32*3*512 = 49152
  if (gid >= 32 * 3 * 512) return;
  int e    = gid & 15;
  int lane = (gid >> 4) & 31;
  int frag = gid >> 9;                 // t*3 + kc
  int kc = frag % 3, t = frag / 3;
  int m = t * 16 + (lane & 15);
  int K = kc * 32 + ((lane < 16) ? 0 : 16) + e;
  dst[gid] = (_Float16)(w[m * CH + K] * rw[m]);
}

// ---------------------------------------------------------------------------
// Prep: pack a [nOut x kIn] fp32 weight (reference layout: out-major) into
// B-fragment layout for D = A(MxK) * B(KxN):  B[k][n] = w[n][k], K padded.
// 16-bit B layout (32x16): lane<16: N=lane, halves e<8 -> K=e, e>=8 -> K=e+8;
//                          lane>=16: N=lane-16, halves e<8 -> K=8+e, e>=8 -> K=16+e.
// ---------------------------------------------------------------------------
__global__ void k_pack_B(const float* __restrict__ w, _Float16* __restrict__ dst,
                         int nOut, int kIn, int total) {
  int gid = blockIdx.x * blockDim.x + threadIdx.x;
  if (gid >= total) return;
  int NT   = nOut >> 4;
  int e    = gid & 15;
  int lane = (gid >> 4) & 31;
  int frag = gid >> 9;
  int nt = frag % NT, kc = frag / NT;
  int K = kc * 32 + ((lane < 16) ? 0 : 8) + ((e < 8) ? e : (e + 8));
  int N = nt * 16 + (lane & 15);
  float v = (K < kIn) ? w[N * kIn + K] : 0.f;
  dst[gid] = (_Float16)v;
}

// ---------------------------------------------------------------------------
// Kernel 1: fused mask GEMM (512x96 @ 96x144 per WG) + normalize + argmax +
// gather/integrate. 288 threads = 9 waves; wave w owns N-tile w (16 columns),
// keeps B fragments resident and loops over 32 M-tiles of the codebook.
// Argmax is done on UN-normalized y (positive per-lane scale 1/||x|| cannot
// change the winner) with a pairwise tree to keep the VALU chain shallow so
// it co-executes with the XDL WMMAs.
// ---------------------------------------------------------------------------
__global__ __launch_bounds__(288) void k_mask(
    const float* __restrict__ x, const float* __restrict__ w_mask,
    const float* __restrict__ rwNorm, const _Float16* __restrict__ wA,
    _Float16* __restrict__ inp) {
  __shared__ alignas(32) _Float16 sPatch[NCOLS * CH];  // [col][c] f16
  __shared__ float sRcpXn[NCOLS];
  __shared__ float sWVal[9 * 32];
  __shared__ int   sWIdx[9 * 32];
  __shared__ float sColVal[NCOLS];
  __shared__ int   sColIdx[NCOLS];
  __shared__ float sMaxSim[PB];
  __shared__ int   sBestV[PB];
  __shared__ int   sBestPos[PB];

  const int tid    = threadIdx.x;
  const int nlBase = blockIdx.x * PB;

  // ---- stage 16 patches to LDS as f16 [col][96] ----
  for (int idx = tid; idx < NCOLS * CH; idx += 288) {
    int col = idx % NCOLS;
    int c   = idx / NCOLS;
    int nl  = nlBase + col / 9;
    int pos = col % 9;
    int n = nl >> 10, lh = (nl >> 5) & 31, lw = nl & 31;
    int i = pos / 3, j = pos % 3;
    sPatch[col * CH + c] =
        (_Float16)x[(((size_t)n * CH + c) * 96 + lh * 3 + i) * 96 + lw * 3 + j];
  }
  __syncthreads();
  if (tid < NCOLS) {
    float s = 0.f;
    for (int c = 0; c < CH; ++c) { float v = (float)sPatch[tid * CH + c]; s += v * v; }
    sRcpXn[tid] = rsqrtf(s);
  }
  __syncthreads();

  // ---- WMMA GEMM + running argmax (y never materialized) ----
  const int lane = tid & 31;
  const int wv   = tid >> 5;                 // 0..8 : N-tile
  const int col  = wv * 16 + (lane & 15);
  const _Float16* bp = &sPatch[col * CH];
  const int bsel = (lane < 16) ? 0 : 16;
  v16h b0 = *(const v16h*)(bp + 0 * 32 + bsel);
  v16h b1 = *(const v16h*)(bp + 1 * 32 + bsel);
  v16h b2 = *(const v16h*)(bp + 2 * 32 + bsel);
  const int rowOff = (lane < 16) ? 0 : 8;    // C-fragment row half select
  float bestVal = -1e30f; int bestIdx = 0;
  for (int t = 0; t < 32; ++t) {
    v16h a0 = *(const v16h*)(wA + ((size_t)(t * 3 + 0) * 32 + lane) * 16);
    v16h a1 = *(const v16h*)(wA + ((size_t)(t * 3 + 1) * 32 + lane) * 16);
    v16h a2 = *(const v16h*)(wA + ((size_t)(t * 3 + 2) * 32 + lane) * 16);
    v8f acc = {};
    acc = __builtin_amdgcn_wmma_f32_16x16x32_f16(false, a0, false, b0, (short)0, acc, false, false);
    acc = __builtin_amdgcn_wmma_f32_16x16x32_f16(false, a1, false, b1, (short)0, acc, false, false);
    acc = __builtin_amdgcn_wmma_f32_16x16x32_f16(false, a2, false, b2, (short)0, acc, false, false);
    // 3-level pairwise argmax tree over the 8 fragment rows (shallow chain),
    // then one merge into the running best. No normalization in the loop.
    float v01 = acc[0]; int i01 = 0;
    if (acc[1] > v01) { v01 = acc[1]; i01 = 1; }
    float v23 = acc[2]; int i23 = 2;
    if (acc[3] > v23) { v23 = acc[3]; i23 = 3; }
    float v45 = acc[4]; int i45 = 4;
    if (acc[5] > v45) { v45 = acc[5]; i45 = 5; }
    float v67 = acc[6]; int i67 = 6;
    if (acc[7] > v67) { v67 = acc[7]; i67 = 7; }
    float va = v01; int ia = i01;
    if (v23 > va) { va = v23; ia = i23; }
    float vb = v45; int ib = i45;
    if (v67 > vb) { vb = v67; ib = i67; }
    if (vb > va) { va = vb; ia = ib; }
    if (va > bestVal) { bestVal = va; bestIdx = t * 16 + rowOff + ia; }
  }
  sWVal[wv * 32 + lane] = bestVal * sRcpXn[col];   // normalize once at the end
  sWIdx[wv * 32 + lane] = bestIdx;
  __syncthreads();

  // per-column reduce (lane pair l / l+16 share column)
  if (tid < NCOLS) {
    int nt = tid >> 4, l = tid & 15;
    float v0 = sWVal[nt * 32 + l], v1 = sWVal[nt * 32 + l + 16];
    sColVal[tid] = (v0 >= v1) ? v0 : v1;
    sColIdx[tid] = (v0 >= v1) ? sWIdx[nt * 32 + l] : sWIdx[nt * 32 + l + 16];
  }
  __syncthreads();
  // per-patch argmax over 9 positions
  if (tid < PB) {
    float bv = -1e30f; int bvc = 0, bpos = 0;
    for (int pos = 0; pos < 9; ++pos) {
      float v = sColVal[tid * 9 + pos];
      if (v > bv) { bv = v; bvc = sColIdx[tid * 9 + pos]; bpos = pos; }
    }
    sMaxSim[tid] = bv; sBestV[tid] = bvc; sBestPos[tid] = bpos;
  }
  __syncthreads();

  // sim_map of winning codebook row (recomputed, cheap) -> inp[96..104]
  if (tid < NCOLS) {
    int p = tid / 9, pos = tid % 9;
    int vv = sBestV[p];
    float dot = 0.f;
    for (int c = 0; c < CH; ++c)
      dot += w_mask[vv * CH + c] * (float)sPatch[(p * 9 + pos) * CH + c];
    float sim = dot * rwNorm[vv] * sRcpXn[p * 9 + pos];
    inp[(size_t)(nlBase + p) * 128 + 96 + pos] = (_Float16)sim;
  }
  // integrate = max_act*s + vc_vec*(1-s) -> inp[0..95]
  for (int idx = tid; idx < PB * CH; idx += 288) {
    int p = idx / CH, c = idx % CH;
    float s   = sMaxSim[p];
    float act = (float)sPatch[(p * 9 + sBestPos[p]) * CH + c];
    float vcv = w_mask[sBestV[p] * CH + c];
    inp[(size_t)(nlBase + p) * 128 + c] = (_Float16)(act * s + vcv * (1.f - s));
  }
  // zero pad K 105..127
  for (int idx = tid; idx < PB * 23; idx += 288) {
    inp[(size_t)(nlBase + idx / 23) * 128 + 105 + idx % 23] = (_Float16)0.f;
  }
}

// ---------------------------------------------------------------------------
// Fused MLP kernel helpers
// ---------------------------------------------------------------------------
__device__ inline v16h load_a(const _Float16* act, int Kpad, int kc, int lane) {
  int row  = lane & 15;
  int base = kc * 32 + ((lane < 16) ? 0 : 8);
  const _Float16* p = act + row * Kpad + base;
  v8h lo = *(const v8h*)(p);
  v8h hi = *(const v8h*)(p + 16);
  v16h a;
#pragma unroll
  for (int e = 0; e < 8; ++e) { a[e] = lo[e]; a[e + 8] = hi[e]; }
  return a;
}

// activ: 0=none, 1=leaky_relu(0.2), 2=tanh, 3=relu
__device__ inline void layer(const _Float16* actIn, int Kpad,
                             const _Float16* __restrict__ wB,
                             const float* __restrict__ bias, int NT,
                             _Float16* actOut, int KpadOut,
                             int activ, int lane, int wv) {
  const int KC = Kpad >> 5;
  for (int nt = wv; nt < NT; nt += 8) {
    v8f acc = {};
    for (int kc = 0; kc < KC; ++kc) {
      v16h a = load_a(actIn, Kpad, kc, lane);
      v16h b = *(const v16h*)(wB + ((size_t)(kc * NT + nt) * 32 + lane) * 16);
      acc = __builtin_amdgcn_wmma_f32_16x16x32_f16(false, a, false, b, (short)0, acc, false, false);
    }
    float bv = bias[nt * 16 + (lane & 15)];
#pragma unroll
    for (int r = 0; r < 8; ++r) {
      float z = acc[r] + bv;
      if (activ == 1)      z = (z < 0.f) ? 0.2f * z : z;
      else if (activ == 2) z = tanhf(z);
      else if (activ == 3) z = fmaxf(z, 0.f);
      int prow = r + ((lane < 16) ? 0 : 8);
      actOut[prow * KpadOut + nt * 16 + (lane & 15)] = (_Float16)z;
    }
  }
}

// ---------------------------------------------------------------------------
// Kernel 2: fused 5-layer MLP + 1x1 conv + fold. 16 patches / WG, 8 waves.
// ---------------------------------------------------------------------------
__global__ __launch_bounds__(256) void k_mlp(
    const _Float16* __restrict__ inp,
    const _Float16* __restrict__ wi1B, const float* __restrict__ b_i1,
    const _Float16* __restrict__ wi2B, const float* __restrict__ b_i2,
    const _Float16* __restrict__ wl1B, const float* __restrict__ b_l1,
    const _Float16* __restrict__ wl2B, const float* __restrict__ b_l2,
    const _Float16* __restrict__ wcB,
    float* __restrict__ out) {
  __shared__ alignas(32) _Float16 bufA[4608];   // inp(16x128)/vcs(16x96)/h2T(144x32)
  __shared__ alignas(32) _Float16 bufB[4608];   // h(16x288)/h1(16x160)
  const int tid = threadIdx.x, lane = tid & 31, wv = tid >> 5;
  const int nlBase = blockIdx.x * PB;

  for (int idx = tid; idx < PB * 128; idx += 256)
    bufA[idx] = inp[(size_t)nlBase * 128 + idx];
  __syncthreads();

  layer(bufA, 128, wi1B, b_i1, 18, bufB, 288, /*leaky*/1, lane, wv);
  __syncthreads();
  layer(bufB, 288, wi2B, b_i2, 6, bufA, 96, /*tanh*/2, lane, wv);
  __syncthreads();
  layer(bufA, 96, wl1B, b_l1, 9, bufB, 160, /*relu*/3, lane, wv);
  for (int idx = tid; idx < PB * 16; idx += 256)          // zero K-pad 144..159
    bufB[(idx / 16) * 160 + 144 + (idx & 15)] = (_Float16)0.f;
  __syncthreads();

  // Layer 4: h2 = h1 @ w_l2^T + b ; scatter into [144 rows=(p,pos)][32 ch]
  for (int nt = wv; nt < 18; nt += 8) {
    v8f acc = {};
    for (int kc = 0; kc < 5; ++kc) {
      v16h a = load_a(bufB, 160, kc, lane);
      v16h b = *(const v16h*)(wl2B + ((size_t)(kc * 18 + nt) * 32 + lane) * 16);
      acc = __builtin_amdgcn_wmma_f32_16x16x32_f16(false, a, false, b, (short)0, acc, false, false);
    }
    int feat = nt * 16 + (lane & 15);
    float bv = b_l2[feat];
    int cc = feat / 9, pos = feat % 9;
#pragma unroll
    for (int r = 0; r < 8; ++r) {
      int p = r + ((lane < 16) ? 0 : 8);
      bufA[(p * 9 + pos) * 32 + cc] = (_Float16)(acc[r] + bv);
    }
  }
  __syncthreads();

  // Layer 5: out[(p,pos), o] = h2T @ w_conv^T ; fold-scatter to [n,c,H,W]
  for (int tile = wv; tile < 54; tile += 8) {
    int mt = tile / 6, nt = tile % 6;
    v16h a = load_a(bufA + mt * 16 * 32, 32, 0, lane);
    v16h b = *(const v16h*)(wcB + ((size_t)nt * 32 + lane) * 16);
    v8f acc = {};
    acc = __builtin_amdgcn_wmma_f32_16x16x32_f16(false, a, false, b, (short)0, acc, false, false);
    int o = nt * 16 + (lane & 15);
#pragma unroll
    for (int r = 0; r < 8; ++r) {
      int rg = mt * 16 + r + ((lane < 16) ? 0 : 8);
      int p = rg / 9, pos = rg % 9;
      int nl = nlBase + p;
      int n = nl >> 10, lh = (nl >> 5) & 31, lw = nl & 31;
      int i = pos / 3, j = pos % 3;
      out[(((size_t)n * CH + o) * 96 + lh * 3 + i) * 96 + lw * 3 + j] = acc[r];
    }
  }
}

// ---------------------------------------------------------------------------
// Host launch
// ---------------------------------------------------------------------------
extern "C" void kernel_launch(void* const* d_in, const int* in_sizes, int n_in,
                              void* d_out, int out_size, void* d_ws, size_t ws_size,
                              hipStream_t stream) {
  (void)in_sizes; (void)n_in; (void)out_size; (void)ws_size;
  const float* x      = (const float*)d_in[0];
  const float* w_mask = (const float*)d_in[1];
  const float* w_i1   = (const float*)d_in[2];
  const float* b_i1   = (const float*)d_in[3];
  const float* w_i2   = (const float*)d_in[4];
  const float* b_i2   = (const float*)d_in[5];
  const float* w_l1   = (const float*)d_in[6];
  const float* b_l1   = (const float*)d_in[7];
  const float* w_l2   = (const float*)d_in[8];
  const float* b_l2   = (const float*)d_in[9];
  const float* w_conv = (const float*)d_in[10];
  float* out = (float*)d_out;

  char* ws = (char*)d_ws;
  size_t off = 0;
  auto carve = [&](size_t bytes) {
    void* p = ws + off;
    off = (off + bytes + 255) & ~(size_t)255;
    return p;
  };
  float*    rwNorm = (float*)   carve((size_t)VC * 4);
  _Float16* wA     = (_Float16*)carve((size_t)32 * 3 * 512 * 2);
  _Float16* inp    = (_Float16*)carve((size_t)NL_TOT * 128 * 2);
  _Float16* wi1B   = (_Float16*)carve((size_t)4 * 18 * 512 * 2);
  _Float16* wi2B   = (_Float16*)carve((size_t)9 * 6 * 512 * 2);
  _Float16* wl1B   = (_Float16*)carve((size_t)3 * 9 * 512 * 2);
  _Float16* wl2B   = (_Float16*)carve((size_t)5 * 18 * 512 * 2);
  _Float16* wcB    = (_Float16*)carve((size_t)1 * 6 * 512 * 2);

  k_wnorm<<<2, 256, 0, stream>>>(w_mask, rwNorm);
  k_pack_wmaskA<<<(32 * 3 * 512 + 255) / 256, 256, 0, stream>>>(w_mask, rwNorm, wA);
  k_pack_B<<<(4 * 18 * 512 + 255) / 256, 256, 0, stream>>>(w_i1, wi1B, 288, 105, 4 * 18 * 512);
  k_pack_B<<<(9 * 6 * 512 + 255) / 256, 256, 0, stream>>>(w_i2, wi2B, 96, 288, 9 * 6 * 512);
  k_pack_B<<<(3 * 9 * 512 + 255) / 256, 256, 0, stream>>>(w_l1, wl1B, 144, 96, 3 * 9 * 512);
  k_pack_B<<<(5 * 18 * 512 + 255) / 256, 256, 0, stream>>>(w_l2, wl2B, 288, 144, 5 * 18 * 512);
  k_pack_B<<<(1 * 6 * 512 + 255) / 256, 256, 0, stream>>>(w_conv, wcB, 96, 32, 1 * 6 * 512);

  k_mask<<<NL_TOT / PB, 288, 0, stream>>>(x, w_mask, rwNorm, wA, inp);
  k_mlp<<<NL_TOT / PB, 256, 0, stream>>>(inp, wi1B, b_i1, wi2B, b_i2,
                                         wl1B, b_l1, wl2B, b_l2, wcB, out);
}
